// GaitGraphNet_49555332661655
// MI455X (gfx1250) — matrix operand order repaired and checked by reading, not compiled, not run on hardware
//
#include <hip/hip_runtime.h>
#include <hip/hip_bf16.h>
#include <math.h>

// ---------------- Problem constants (match reference) ----------------
#define BB   128          // batch
#define TT   256          // sequence length
#define NN   17           // joints
#define FF   64           // gcn feature dim
#define DD   384          // transformer d_model
#define HH   8            // heads
#define HD   48           // head dim
#define DFF  768          // feedforward
#define GG   (BB*TT)      // 32768 tokens
#define VTS  264          // padded V^T row stride (bank-spread, 16B aligned)

typedef __attribute__((ext_vector_type(16))) __bf16  v16bf;
typedef __attribute__((ext_vector_type(8)))  float   v8f;
typedef __attribute__((ext_vector_type(8)))  unsigned v8u;

// Pack two f32 -> packed bf16 dword (round via +0x8000, one v_perm_b32).
static __device__ __forceinline__ unsigned pack_bf16(float lo, float hi) {
    unsigned a = __builtin_bit_cast(unsigned, lo) + 0x8000u;
    unsigned b = __builtin_bit_cast(unsigned, hi) + 0x8000u;
    return __builtin_amdgcn_perm(b, a, 0x07060302u);  // {hi[31:16], lo[31:16]}
}
static __device__ __forceinline__ unsigned short f2bf_bits(float f) {
    return (unsigned short)((__builtin_bit_cast(unsigned, f) + 0x8000u) >> 16);
}

// ================================================================
// One-time weight conversion: out[n*K+k] = bf16(W[n*wsn + k*wsk]).
// Folds transposes; all GEMM B-operands become contiguous bf16 rows.
// ================================================================
__global__ void convert_w_kernel(const float* __restrict__ W, unsigned short* __restrict__ out,
                                 int total, int K, int wsn, int wsk) {
    const int idx = blockIdx.x * 256 + threadIdx.x;
    if (idx >= total) return;
    const int n = idx / K, k = idx % K;
    out[idx] = f2bf_bits(W[(size_t)n * wsn + (size_t)k * wsk]);
}

// ================================================================
// GEMM: C[M,N] = act( A[M,K](f32) @ Wb16^T + bias ), Wb16 row-major [N,K] bf16.
// bf16 WMMA 16x16x32, fp32 accumulate. Block = 256 thr = 8 waves.
// Each wave: 16 rows x (NT*16) cols; A fragment converted once and held in
// regs across NT column tiles. B fragment = one aligned 32B bf16 load.
// M must be a multiple of 128.
// ================================================================
template <int NT>
__global__ void gemm_bf16(const float* __restrict__ A, int lda,
                          const unsigned short* __restrict__ Wb16,
                          const float* __restrict__ bias,
                          float* __restrict__ C, int ldc,
                          int M, int K, int relu) {
    const int lane = threadIdx.x & 31;
    const int wave = threadIdx.x >> 5;          // 0..7
    const int hf   = (lane >> 4) & 1;
    const int r    = lane & 15;
    const int m0   = blockIdx.x * 128 + wave * 16;
    const int nb   = blockIdx.y * (NT * 16);
    const int arow = m0 + r;
    const float* Abase = A + (size_t)arow * lda + hf * 16;

    v8f acc[NT];
#pragma unroll
    for (int t = 0; t < NT; ++t) acc[t] = v8f{};

    for (int kk = 0; kk < K; kk += 32) {
        __builtin_prefetch(Abase + kk + 64, 0, 3);
        // ---- A fragment: K(e) = kk + hf*16 + e
        v8u ua;
#pragma unroll
        for (int j = 0; j < 8; ++j) {
            const int off = kk + 2 * j;
            const float2 v2 = *(const float2*)(Abase + off);
            ua[j] = pack_bf16(v2.x, v2.y);
        }
        const v16bf ta = __builtin_bit_cast(v16bf, ua);
        // ---- NT column tiles reuse the A fragment; B = direct bf16 load
#pragma unroll
        for (int t = 0; t < NT; ++t) {
            const int bcol = nb + t * 16 + r;
            const v8u ub = *(const v8u*)(Wb16 + (size_t)bcol * K + kk + hf * 16);
            const v16bf tb = __builtin_bit_cast(v16bf, ub);
            acc[t] = __builtin_amdgcn_wmma_f32_16x16x32_bf16(false, ta, false, tb,
                                                             (short)0, acc[t], false, false);
        }
    }
#pragma unroll
    for (int t = 0; t < NT; ++t) {
        const int col = nb + t * 16 + r;
        const float bs = bias ? bias[col] : 0.f;
#pragma unroll
        for (int v = 0; v < 8; ++v) {
            const int row = m0 + v + hf * 8;
            float c = acc[t][v] + bs;
            if (relu) c = fmaxf(c, 0.f);
            C[(size_t)row * ldc + col] = c;
        }
    }
}

// ================================================================
// GCN: normalization precompute (tiny, serial)
// ================================================================
__global__ void gcn_norm_kernel(const float* __restrict__ ew,
                                const int* __restrict__ eidx,
                                float* __restrict__ norm,
                                float* __restrict__ nself, int E) {
    if (threadIdx.x == 0 && blockIdx.x == 0) {
        const int* src = eidx;
        const int* dst = eidx + E;
        float deg[NN];
        for (int n = 0; n < NN; ++n) deg[n] = 1.f;       // self-loop fill 1.0
        for (int e = 0; e < E; ++e) deg[dst[e]] += ew[e];
        float dis[NN];
        for (int n = 0; n < NN; ++n) dis[n] = rsqrtf(deg[n]);
        for (int e = 0; e < E; ++e) norm[e] = dis[src[e]] * ew[e] * dis[dst[e]];
        for (int n = 0; n < NN; ++n) nself[n] = dis[n] * dis[n];
    }
}

// ================================================================
// GCN message passing + bias + relu + residual + mean/max readout.
// One block per graph, 64 threads = feature channels.
// ================================================================
__global__ void gcn_msg_kernel(const float* __restrict__ hw,
                               const float* __restrict__ h_in,
                               float* __restrict__ h_out,
                               const float* __restrict__ gb,
                               const int* __restrict__ src,
                               const int* __restrict__ dst,
                               const float* __restrict__ norm,
                               const float* __restrict__ nself,
                               float* __restrict__ z, int E, int layer) {
    const int g = blockIdx.x;
    const int f = threadIdx.x;                  // 0..63
    const float* hwg = hw   + (size_t)g * NN * FF;
    const float* hg  = h_in + (size_t)g * NN * FF;
    float*       ho  = h_out + (size_t)g * NN * FF;
    float out[NN];
#pragma unroll
    for (int n = 0; n < NN; ++n) out[n] = hwg[n * FF + f] * nself[n];
    for (int e = 0; e < E; ++e) out[dst[e]] += hwg[src[e] * FF + f] * norm[e];
    const float bias = gb[f];
    float mean = 0.f, mx = -INFINITY;
#pragma unroll
    for (int n = 0; n < NN; ++n) {
        const float v = fmaxf(out[n] + bias, 0.f) + hg[n * FF + f];
        ho[n * FF + f] = v;
        mean += v;
        mx = fmaxf(mx, v);
    }
    z[(size_t)g * DD + layer * 128 + f]      = mean * (1.f / (float)NN);
    z[(size_t)g * DD + layer * 128 + 64 + f] = mx;
}

// ================================================================
// Positional encoding add (analytic)
// ================================================================
__global__ void pe_kernel(float* __restrict__ z) {
    const size_t idx = (size_t)blockIdx.x * blockDim.x + threadIdx.x;  // GG*DD
    const int d = (int)(idx % DD);
    const int t = (int)((idx / DD) % TT);
    const float i2   = (float)((d >> 1) << 1);
    const float freq = __expf(-i2 * (9.210340371976184f / (float)DD)); // ln(10000)
    const float ang  = (float)t * freq;
    z[idx] += (d & 1) ? __cosf(ang) : __sinf(ang);
}

// ================================================================
// Fused attention for one (b, head, 64-row stripe), 128 thr = 4 waves.
// LDS: 64x256 f32 scores + 48x264 bf16 V^T (staged once at entry).
// Phase1: S = Q K^T/sqrt(hd)+mask (WMMA, Q frags hoisted) -> LDS
// Phase2: float4 row softmax in LDS
// Phase3: O = P V (WMMA; P frag from LDS once per K-step, V frags are
//         two ds_load_b128 from the staged V^T)
// ================================================================
__global__ void attn_kernel(const float* __restrict__ qkv,
                            const unsigned char* __restrict__ mask,
                            float* __restrict__ o) {
    extern __shared__ float smem[];
    float* sS = smem;                                       // 64*256 f32
    unsigned short* svt = (unsigned short*)(smem + 64 * TT); // 48*264 bf16
    const int bid = blockIdx.x;
    const int b   = bid / (HH * 4);
    const int rem = bid % (HH * 4);
    const int h   = rem >> 2;
    const int rb  = rem & 3;
    const int lane = threadIdx.x & 31;
    const int wave = threadIdx.x >> 5;          // 0..3
    const int hf   = (lane >> 4) & 1;
    const int r    = lane & 15;
    const int m_base = rb * 64;
    const float scale = 0.14433756729740643f;   // 1/sqrt(48)
    const float* qbase = qkv + (size_t)b * TT * (3 * DD) + h * HD;
    const float* vbase = qbase + 2 * DD;

    // ---- stage V^T (bf16) into LDS ----
    for (int i = threadIdx.x; i < TT * HD; i += 128) {
        const int t = i / HD, d = i % HD;
        svt[d * VTS + t] = f2bf_bits(vbase[(size_t)t * (3 * DD) + d]);
    }

    // ---- hoist Q fragments (depend on kk only) ----
    v16bf qa[2];
    const int arow = m_base + wave * 16 + r;
#pragma unroll
    for (int ki = 0; ki < 2; ++ki) {
        const int kk = ki * 32;
        v8u ua;
#pragma unroll
        for (int j = 0; j < 8; ++j) {
            const int ka = kk + hf * 16 + 2 * j;
            float2 v2 = {0.f, 0.f};
            if (ka < HD) v2 = *(const float2*)(qbase + (size_t)arow * (3 * DD) + ka);
            ua[j] = pack_bf16(v2.x, v2.y);
        }
        qa[ki] = __builtin_bit_cast(v16bf, ua);
    }

    // ---- phase 1: scores stripe ----
    for (int nt = 0; nt < 16; ++nt) {
        const int n0 = nt * 16;
        const int brow = n0 + r;                // key time index
        v8f acc = {};
#pragma unroll
        for (int ki = 0; ki < 2; ++ki) {
            const int kk = ki * 32;
            v8u ub;
#pragma unroll
            for (int j = 0; j < 8; ++j) {
                const int kb = kk + hf * 16 + 2 * j;
                float2 v2 = {0.f, 0.f};
                if (kb < HD) v2 = *(const float2*)(qbase + (size_t)brow * (3 * DD) + DD + kb);
                ub[j] = pack_bf16(v2.x, v2.y);
            }
            const v16bf tb = __builtin_bit_cast(v16bf, ub);
            acc = __builtin_amdgcn_wmma_f32_16x16x32_bf16(false, qa[ki], false, tb,
                                                          (short)0, acc, false, false);
        }
        const float mb = mask[(size_t)b * TT + n0 + r] ? -INFINITY : 0.f;
#pragma unroll
        for (int v = 0; v < 8; ++v) {
            const int lr = wave * 16 + v + hf * 8;
            sS[lr * TT + n0 + r] = acc[v] * scale + mb;
        }
    }
    __syncthreads();   // orders phase1 stores AND V^T staging before use

    // ---- phase 2: softmax, one thread per row, float4 LDS traffic ----
    if (threadIdx.x < 64) {
        float4* row4 = (float4*)(sS + threadIdx.x * TT);
        float mx = -INFINITY;
        for (int j = 0; j < TT / 4; ++j) {
            const float4 v = row4[j];
            mx = fmaxf(mx, fmaxf(fmaxf(v.x, v.y), fmaxf(v.z, v.w)));
        }
        float s = 0.f;
        for (int j = 0; j < TT / 4; ++j) {
            float4 v = row4[j];
            v.x = __expf(v.x - mx); v.y = __expf(v.y - mx);
            v.z = __expf(v.z - mx); v.w = __expf(v.w - mx);
            row4[j] = v;
            s += (v.x + v.y) + (v.z + v.w);
        }
        const float inv = 1.f / s;
        for (int j = 0; j < TT / 4; ++j) {
            float4 v = row4[j];
            v.x *= inv; v.y *= inv; v.z *= inv; v.w *= inv;
            row4[j] = v;
        }
    }
    __syncthreads();

    // ---- phase 3: O = P @ V ----
    v8f acc3[3];
#pragma unroll
    for (int t = 0; t < 3; ++t) acc3[t] = v8f{};
    const int prow = wave * 16 + r;             // local row in LDS stripe
    for (int kk = 0; kk < TT; kk += 32) {
        v8u ua;
#pragma unroll
        for (int j = 0; j < 8; ++j) {
            const int ka = kk + hf * 16 + 2 * j;
            const float2 v2 = *(const float2*)(sS + prow * TT + ka);
            ua[j] = pack_bf16(v2.x, v2.y);
        }
        const v16bf pa = __builtin_bit_cast(v16bf, ua);
#pragma unroll
        for (int dt = 0; dt < 3; ++dt) {
            const int dcol = dt * 16 + r;       // 0..47, always valid (48 = 3*16)
            const unsigned short* vp = svt + dcol * VTS + kk + hf * 16;
            const uint4 lo = *(const uint4*)(vp);
            const uint4 hi = *(const uint4*)(vp + 8);
            v8u ub;
            ub[0] = lo.x; ub[1] = lo.y; ub[2] = lo.z; ub[3] = lo.w;
            ub[4] = hi.x; ub[5] = hi.y; ub[6] = hi.z; ub[7] = hi.w;
            const v16bf tb = __builtin_bit_cast(v16bf, ub);
            acc3[dt] = __builtin_amdgcn_wmma_f32_16x16x32_bf16(false, pa, false, tb,
                                                               (short)0, acc3[dt], false, false);
        }
    }
#pragma unroll
    for (int dt = 0; dt < 3; ++dt) {
#pragma unroll
        for (int v = 0; v < 8; ++v) {
            const int t = m_base + wave * 16 + v + hf * 8;
            o[((size_t)b * TT + t) * DD + h * HD + dt * 16 + r] = acc3[dt][v];
        }
    }
}

// ================================================================
// z = LayerNorm(z + res) * g + b ; one wave per row of 384
// ================================================================
__global__ void addln_kernel(float* __restrict__ z, const float* __restrict__ res,
                             const float* __restrict__ g, const float* __restrict__ bt,
                             int Mrows) {
    const int lane = threadIdx.x & 31;
    const int wave = threadIdx.x >> 5;
    const int row  = blockIdx.x * 8 + wave;
    if (row >= Mrows) return;
    float* zr = z + (size_t)row * DD;
    const float* rr = res + (size_t)row * DD;
    float x[12];
    float s = 0.f;
#pragma unroll
    for (int i = 0; i < 12; ++i) { x[i] = zr[lane + i * 32] + rr[lane + i * 32]; s += x[i]; }
#pragma unroll
    for (int off = 16; off > 0; off >>= 1) s += __shfl_xor(s, off);
    const float m = s * (1.f / (float)DD);
    float vs = 0.f;
#pragma unroll
    for (int i = 0; i < 12; ++i) { const float d = x[i] - m; vs += d * d; }
#pragma unroll
    for (int off = 16; off > 0; off >>= 1) vs += __shfl_xor(vs, off);
    const float inv = rsqrtf(vs * (1.f / (float)DD) + 1e-5f);
#pragma unroll
    for (int i = 0; i < 12; ++i) {
        const int d = lane + i * 32;
        zr[d] = (x[i] - m) * inv * g[d] + bt[d];
    }
}

// ================================================================
// Mean pool over T
// ================================================================
__global__ void pool_kernel(const float* __restrict__ z, float* __restrict__ pooled) {
    const int b = blockIdx.x;
    const int d = threadIdx.x;                  // 384
    float s = 0.f;
    for (int t = 0; t < TT; ++t) s += z[((size_t)b * TT + t) * DD + d];
    pooled[b * DD + d] = s * (1.f / (float)TT);
}

// ================================================================
// Head tail: BN(eval) + relu + final [256 -> 2] linear
// ================================================================
__global__ void head_kernel(const float* __restrict__ p1,
                            const float* __restrict__ bng, const float* __restrict__ bnb,
                            const float* __restrict__ bnm, const float* __restrict__ bnv,
                            const float* __restrict__ pw2, const float* __restrict__ pb2,
                            float* __restrict__ out) {
    const int tid = threadIdx.x;                // 256 = 128 rows * 2 outputs
    const int b = tid >> 1, j = tid & 1;
    float s = 0.f;
    for (int c = 0; c < 256; ++c) {
        float v = p1[b * 256 + c];
        v = (v - bnm[c]) * rsqrtf(bnv[c] + 1e-5f) * bng[c] + bnb[c];
        v = fmaxf(v, 0.f);
        s += v * pw2[j * 256 + c];
    }
    out[b * 2 + j] = s + pb2[j];
}

// ================================================================
// Host orchestration
// ================================================================
extern "C" void kernel_launch(void* const* d_in, const int* in_sizes, int n_in,
                              void* d_out, int out_size, void* d_ws, size_t ws_size,
                              hipStream_t stream) {
    (void)n_in; (void)out_size; (void)ws_size;
    const float* x    = (const float*)d_in[0];
    const float* ew   = (const float*)d_in[1];
    const float* gcnw = (const float*)d_in[2];
    const float* gcnb = (const float*)d_in[3];
    const float* ipw  = (const float*)d_in[4];
    const float* ipb  = (const float*)d_in[5];
    const float* opw  = (const float*)d_in[6];
    const float* opb  = (const float*)d_in[7];
    const float* l1w  = (const float*)d_in[8];
    const float* l1b  = (const float*)d_in[9];
    const float* l2w  = (const float*)d_in[10];
    const float* l2b  = (const float*)d_in[11];
    const float* ln1g = (const float*)d_in[12];
    const float* ln1b = (const float*)d_in[13];
    const float* ln2g = (const float*)d_in[14];
    const float* ln2b = (const float*)d_in[15];
    const float* pw1  = (const float*)d_in[16];
    const float* pb1  = (const float*)d_in[17];
    const float* bng  = (const float*)d_in[18];
    const float* bnb  = (const float*)d_in[19];
    const float* bnm  = (const float*)d_in[20];
    const float* bnv  = (const float*)d_in[21];
    const float* pw2  = (const float*)d_in[22];
    const float* pb2  = (const float*)d_in[23];
    const int*   eidx = (const int*)d_in[24];
    const unsigned char* mask = (const unsigned char*)d_in[25];
    const int E = in_sizes[1];                  // 32 directed edges

    // -------- workspace layout (floats), overlaid by live range --------
    float* ws = (float*)d_ws;
    const size_t GN64 = (size_t)GG * NN * FF;   // 35,651,584
    float* f_h   = ws;                          // GCN hidden   [GN64]
    float* f_hw  = ws + GN64;                   // GCN h@W      [GN64]
    float* f_qkv = ws;                          // G*1152 (overlays GCN region)
    float* f_o   = ws + (size_t)GG * 3 * DD;    // attn out  G*384
    float* f_o2  = f_o + (size_t)GG * DD;       // out_proj  G*384
    float* f_ff1 = ws;                          // G*768 (overlays qkv)
    float* f_ff2 = f_o;                         // overlays o
    float* f_z   = ws + 2 * GN64;               // tokens    G*384
    float* f_pl  = f_z + (size_t)GG * DD;       // pooled    128*384
    float* f_p1  = f_pl + (size_t)BB * DD;      // head mid  128*256
    float* f_nrm = f_p1 + (size_t)BB * 256;     // edge norm [E]
    float* f_ns  = f_nrm + 64;                  // self norm [17]
    // ---- bf16 weight region (32B aligned) ----
    size_t woff = (size_t)(f_ns + 32 - ws);
    woff = (woff + 7) & ~(size_t)7;
    unsigned short* wt = (unsigned short*)(ws + woff);
    unsigned short* wt_gcn = wt;                               // 3 * 64*64
    unsigned short* wt_ip  = wt_gcn + 3 * FF * FF;             // 2 * 1152*384
    unsigned short* wt_op  = wt_ip  + 2 * (size_t)3 * DD * DD; // 2 * 384*384
    unsigned short* wt_l1  = wt_op  + 2 * (size_t)DD * DD;     // 2 * 768*384
    unsigned short* wt_l2  = wt_l1  + 2 * (size_t)DFF * DD;    // 2 * 384*768
    unsigned short* wt_p1  = wt_l2  + 2 * (size_t)DD * DFF;    // 256*384

    dim3 blk(256);
    // -------- one-time weight conversions (tiny) --------
    for (int l = 0; l < 3; ++l)   // gcn_w[l] is [K=64, N=64] -> transpose-convert
        convert_w_kernel<<<(FF * FF + 255) / 256, blk, 0, stream>>>(
            gcnw + (size_t)l * FF * FF, wt_gcn + (size_t)l * FF * FF, FF * FF, FF, 1, FF);
    for (int l = 0; l < 2; ++l) {
        convert_w_kernel<<<(3 * DD * DD + 255) / 256, blk, 0, stream>>>(
            ipw + (size_t)l * 3 * DD * DD, wt_ip + (size_t)l * 3 * DD * DD, 3 * DD * DD, DD, DD, 1);
        convert_w_kernel<<<(DD * DD + 255) / 256, blk, 0, stream>>>(
            opw + (size_t)l * DD * DD, wt_op + (size_t)l * DD * DD, DD * DD, DD, DD, 1);
        convert_w_kernel<<<(DFF * DD + 255) / 256, blk, 0, stream>>>(
            l1w + (size_t)l * DFF * DD, wt_l1 + (size_t)l * DFF * DD, DFF * DD, DD, DD, 1);
        convert_w_kernel<<<(DD * DFF + 255) / 256, blk, 0, stream>>>(
            l2w + (size_t)l * DD * DFF, wt_l2 + (size_t)l * DD * DFF, DD * DFF, DFF, DFF, 1);
    }
    convert_w_kernel<<<(256 * DD + 255) / 256, blk, 0, stream>>>(
        pw1, wt_p1, 256 * DD, DD, DD, 1);

    // -------- GCN --------
    gcn_norm_kernel<<<1, 32, 0, stream>>>(ew, eidx, f_nrm, f_ns, E);
    for (int l = 0; l < 3; ++l) {
        const float* hin = (l == 0) ? x : f_h;
        gemm_bf16<4><<<dim3((GG * NN) / 128, 1), blk, 0, stream>>>(
            hin, FF, wt_gcn + (size_t)l * FF * FF, nullptr, f_hw, FF, GG * NN, FF, 0);
        gcn_msg_kernel<<<GG, 64, 0, stream>>>(f_hw, hin, f_h, gcnb + l * FF,
                                              eidx, eidx + E, f_nrm, f_ns, f_z, E, l);
    }

    // -------- positional encoding --------
    pe_kernel<<<(GG * DD) / 256, blk, 0, stream>>>(f_z);

    // -------- transformer encoder --------
    const size_t attn_lds = 64 * TT * sizeof(float) + HD * VTS * sizeof(unsigned short);
    for (int l = 0; l < 2; ++l) {
        const size_t oD = (size_t)l * DD;
        gemm_bf16<8><<<dim3(GG / 128, (3 * DD) / 128), blk, 0, stream>>>(
            f_z, DD, wt_ip + (size_t)l * 3 * DD * DD, ipb + (size_t)l * 3 * DD,
            f_qkv, 3 * DD, GG, DD, 0);
        attn_kernel<<<BB * HH * 4, 128, attn_lds, stream>>>(f_qkv, mask, f_o);
        gemm_bf16<8><<<dim3(GG / 128, DD / 128), blk, 0, stream>>>(
            f_o, DD, wt_op + (size_t)l * DD * DD, opb + oD, f_o2, DD, GG, DD, 0);
        addln_kernel<<<GG / 8, blk, 0, stream>>>(f_z, f_o2, ln1g + oD, ln1b + oD, GG);
        gemm_bf16<8><<<dim3(GG / 128, DFF / 128), blk, 0, stream>>>(
            f_z, DD, wt_l1 + (size_t)l * DFF * DD, l1b + (size_t)l * DFF,
            f_ff1, DFF, GG, DD, 1 /*relu*/);
        gemm_bf16<8><<<dim3(GG / 128, DD / 128), blk, 0, stream>>>(
            f_ff1, DFF, wt_l2 + (size_t)l * DD * DFF, l2b + oD, f_ff2, DD, GG, DFF, 0);
        addln_kernel<<<GG / 8, blk, 0, stream>>>(f_z, f_ff2, ln2g + oD, ln2b + oD, GG);
    }

    // -------- head --------
    pool_kernel<<<BB, DD, 0, stream>>>(f_z, f_pl);
    gemm_bf16<8><<<dim3(1, 2), blk, 0, stream>>>(
        f_pl, DD, wt_p1, pb1, f_p1, 256, BB, DD, 0);
    head_kernel<<<1, 256, 0, stream>>>(f_p1, bng, bnb, bnm, bnv, pw2, pb2, (float*)d_out);
}
